// AlignedGloveLayer_78580721647926
// MI455X (gfx1250) — compile-verified, matrix operand order checked by reading, not run on hardware
//
#include <hip/hip_runtime.h>
#include <hip/hip_bf16.h>

typedef __attribute__((ext_vector_type(16))) __bf16 v16bf;
typedef __attribute__((ext_vector_type(8)))  __bf16 v8bf;
typedef __attribute__((ext_vector_type(8)))  float  v8f;

#define HID 100

// ---------------------------------------------------------------------------
// Init: zero scalar accumulators, set per-row (dist,idx) packed mins to max.
// ---------------------------------------------------------------------------
__global__ void init_kernel(float* __restrict__ acc,
                            unsigned long long* __restrict__ rmin_fx,
                            unsigned long long* __restrict__ rmin_gy,
                            int Kpad) {
    int t = blockIdx.x * blockDim.x + threadIdx.x;
    if (t < 16) acc[t] = 0.0f;
    if (t < Kpad) {
        rmin_fx[t] = ~0ull;
        rmin_gy[t] = ~0ull;
    }
}

// ---------------------------------------------------------------------------
// Fused MLP + round-trip + cycle row-norm. One thread per row.
//   mapped = relu(X@W1+B1)@W2+B2   (stored, N x dmid)
//   rt     = relu(mapped@V1+C1)@V2+C2
//   cyc_acc += || rt - X_row ||
// ---------------------------------------------------------------------------
__global__ void __launch_bounds__(128) mlp_cycle_kernel(
    const float* __restrict__ X,
    const float* __restrict__ W1, const float* __restrict__ B1,
    const float* __restrict__ W2, const float* __restrict__ B2,
    const float* __restrict__ V1, const float* __restrict__ C1,
    const float* __restrict__ V2, const float* __restrict__ C2,
    float* __restrict__ mapped, float* __restrict__ cyc_acc,
    int N, int din, int dmid)
{
    int r = blockIdx.x * blockDim.x + threadIdx.x;
    if (r >= N) return;
    float h[HID];
    float mid[128];
    const float* xr = X + (size_t)r * din;

    for (int j = 0; j < HID; ++j) h[j] = B1[j];
    for (int i = 0; i < din; ++i) {
        float xv = xr[i];
        const float* w = W1 + (size_t)i * HID;
        for (int j = 0; j < HID; ++j) h[j] = fmaf(xv, w[j], h[j]);
    }
    for (int j = 0; j < HID; ++j) h[j] = fmaxf(h[j], 0.0f);

    for (int j = 0; j < dmid; ++j) mid[j] = B2[j];
    for (int i = 0; i < HID; ++i) {
        float hv = h[i];
        const float* w = W2 + (size_t)i * dmid;
        for (int j = 0; j < dmid; ++j) mid[j] = fmaf(hv, w[j], mid[j]);
    }
    float* mr = mapped + (size_t)r * dmid;
    for (int j = 0; j < dmid; ++j) mr[j] = mid[j];

    // round trip through the other MLP
    for (int j = 0; j < HID; ++j) h[j] = C1[j];
    for (int i = 0; i < dmid; ++i) {
        float mv = mid[i];
        const float* w = V1 + (size_t)i * HID;
        for (int j = 0; j < HID; ++j) h[j] = fmaf(mv, w[j], h[j]);
    }
    for (int j = 0; j < HID; ++j) h[j] = fmaxf(h[j], 0.0f);

    float sq = 0.0f;
    for (int j = 0; j < din; ++j) {
        float s = C2[j];
        for (int i = 0; i < HID; ++i) s = fmaf(h[i], V2[(size_t)i * din + j], s);
        float d = s - xr[j];
        sq = fmaf(d, d, sq);
    }
    atomicAdd(cyc_acc, sqrtf(sq));
}

// ---------------------------------------------------------------------------
// Convert target matrix rows to split-bf16 (hi + lo), pad K to 128, and
// compute exact fp32 row norms. Pad rows get huge norm (never win argmin).
// ---------------------------------------------------------------------------
__global__ void __launch_bounds__(128) convert_B_kernel(
    const float* __restrict__ Bsrc, int N, int d, int Npad,
    __bf16* __restrict__ hi, __bf16* __restrict__ lo, float* __restrict__ nb)
{
    int r = blockIdx.x * blockDim.x + threadIdx.x;
    if (r >= Npad) return;
    float sq = 0.0f;
    for (int k = 0; k < 128; ++k) {
        float v = 0.0f;
        if (r < N && k < d) { v = Bsrc[(size_t)r * d + k]; sq = fmaf(v, v, sq); }
        __bf16 h = (__bf16)v;
        hi[(size_t)r * 128 + k] = h;
        lo[(size_t)r * 128 + k] = (__bf16)(v - (float)h);
    }
    nb[r] = (r < N) ? sq : 1.0e30f;
}

// ---------------------------------------------------------------------------
// Gather mapped rows at index_map, convert to split-bf16 A matrices, compute
// exact ||a||^2, and accumulate supervised pair norms (sup_x, sup_y).
// ---------------------------------------------------------------------------
__global__ void __launch_bounds__(128) gather_A_kernel(
    const float* __restrict__ xm, const float* __restrict__ ym,
    const float* __restrict__ xw, const float* __restrict__ yw,
    const int* __restrict__ imap, int K, int Kpad,
    __bf16* __restrict__ Afx_hi, __bf16* __restrict__ Afx_lo, float* __restrict__ na_fx,
    __bf16* __restrict__ Agy_hi, __bf16* __restrict__ Agy_lo, float* __restrict__ na_gy,
    float* __restrict__ acc)
{
    int k = blockIdx.x * blockDim.x + threadIdx.x;
    if (k >= Kpad) return;
    if (k >= K) {
        for (int j = 0; j < 128; ++j) {
            Afx_hi[(size_t)k * 128 + j] = (__bf16)0.0f;
            Afx_lo[(size_t)k * 128 + j] = (__bf16)0.0f;
            Agy_hi[(size_t)k * 128 + j] = (__bf16)0.0f;
            Agy_lo[(size_t)k * 128 + j] = (__bf16)0.0f;
        }
        na_fx[k] = 1.0e30f;
        na_gy[k] = 1.0e30f;
        return;
    }
    int xi = imap[2 * k + 0];
    int yi = imap[2 * k + 1];

    // fx side: a = x_mapped[xi] (dim 120), supervised vs y_weight[yi]
    float sq = 0.0f, sup = 0.0f;
    for (int j = 0; j < 128; ++j) {
        float v = (j < 120) ? xm[(size_t)xi * 120 + j] : 0.0f;
        sq = fmaf(v, v, sq);
        __bf16 h = (__bf16)v;
        Afx_hi[(size_t)k * 128 + j] = h;
        Afx_lo[(size_t)k * 128 + j] = (__bf16)(v - (float)h);
        if (j < 120) { float d = v - yw[(size_t)yi * 120 + j]; sup = fmaf(d, d, sup); }
    }
    na_fx[k] = sq;
    atomicAdd(&acc[2], sqrtf(sup));

    // gy side: a = y_mapped[yi] (dim 100), supervised vs x_weight[xi]
    sq = 0.0f; sup = 0.0f;
    for (int j = 0; j < 128; ++j) {
        float v = (j < 100) ? ym[(size_t)yi * 100 + j] : 0.0f;
        sq = fmaf(v, v, sq);
        __bf16 h = (__bf16)v;
        Agy_hi[(size_t)k * 128 + j] = h;
        Agy_lo[(size_t)k * 128 + j] = (__bf16)(v - (float)h);
        if (j < 100) { float d = v - xw[(size_t)xi * 100 + j]; sup = fmaf(d, d, sup); }
    }
    na_gy[k] = sq;
    atomicAdd(&acc[3], sqrtf(sup));
}

// ---------------------------------------------------------------------------
// Load a 16-wide bf16 WMMA fragment per ISA layout (§7.12.2): lane L holds
// row (L%16); elements 0..7 = k_base..k_base+7, 8..15 = k_base+16.., with
// k_base = (L/16)*8. Two aligned 16B loads per lane. Works for both global
// and LDS row pointers (address space inferred).
// ---------------------------------------------------------------------------
template <typename P>
__device__ inline v16bf load_frag(P rowp, int k0, int kb) {
    v8bf a = *(const v8bf*)(rowp + k0 + kb);
    v8bf b = *(const v8bf*)(rowp + k0 + 16 + kb);
    v16bf f;
#pragma unroll
    for (int j = 0; j < 8; ++j) { f[j] = a[j]; f[j + 8] = b[j]; }
    return f;
}

// ---------------------------------------------------------------------------
// WMMA 1-NN kernel. Workgroup = 256 threads = 8 waves, one M-tile per block.
// A tile (split hi/lo, 8KB) staged once in LDS and shared by all waves.
// Each wave computes 1 M-tile x 4 N-tiles, reusing A fragments across the 4
// B tiles: 12 WMMAs per k-step vs 4 ds + 16 global fragment loads.
// Out-of-range N tiles are CLAMPED (duplicate candidates are harmless under
// atomicMin) so EXEC stays all-ones for WMMA everywhere.
// Split-bf16 compensated dot: ah*bh + ah*bl + al*bh.
// d2 = ||a||^2 + ||b||^2 - 2*dot; per-row (min,argmin) via shfl_xor + packed
// u64 atomicMin.
// ---------------------------------------------------------------------------
__global__ void __launch_bounds__(256) nn_min_kernel(
    const __bf16* __restrict__ Ahi, const __bf16* __restrict__ Alo,
    const __bf16* __restrict__ Bhi, const __bf16* __restrict__ Blo,
    const float* __restrict__ na, const float* __restrict__ nb,
    unsigned long long* __restrict__ rowmin, int nTiles)
{
    __shared__ __bf16 sAhi[16 * 128];
    __shared__ __bf16 sAlo[16 * 128];

    const int tid   = threadIdx.x;
    const int lane  = tid & 31;
    const int wave  = tid >> 5;
    const int mTile = blockIdx.x;

    // cooperative stage of the A tile (hi+lo) into LDS: 16B/thread/matrix
    {
        const __bf16* gHi = Ahi + (size_t)mTile * 16 * 128;
        const __bf16* gLo = Alo + (size_t)mTile * 16 * 128;
        int e = tid * 8;  // 256 threads * 8 bf16 = 2048 = full tile
        *(v8bf*)(sAhi + e) = *(const v8bf*)(gHi + e);
        *(v8bf*)(sAlo + e) = *(const v8bf*)(gLo + e);
    }
    __syncthreads();

    const int r  = lane & 15;
    const int kb = (lane >> 4) * 8;
    const __bf16* sa_hi = sAhi + r * 128;
    const __bf16* sa_lo = sAlo + r * 128;

    // 4 N tiles per wave, clamped at the grid tail
    const int nTile0 = (blockIdx.y * 8 + wave) * 4;
    int nT[4];
    const __bf16* brh[4];
    const __bf16* brl[4];
#pragma unroll
    for (int t = 0; t < 4; ++t) {
        int nt = nTile0 + t;
        nT[t] = (nt < nTiles) ? nt : (nTiles - 1);
        brh[t] = Bhi + (size_t)(nT[t] * 16 + r) * 128;
        brl[t] = Blo + (size_t)(nT[t] * 16 + r) * 128;
    }

    // hint the next n-tile-group's B stream into cache
    __builtin_prefetch(brh[3] + (size_t)4 * 16 * 128, 0, 1);

    v8f acc[4] = {};
#pragma unroll
    for (int k0 = 0; k0 < 128; k0 += 32) {
        v16bf ah = load_frag(sa_hi, k0, kb);
        v16bf al = load_frag(sa_lo, k0, kb);
#pragma unroll
        for (int t = 0; t < 4; ++t) {
            v16bf bh = load_frag(brh[t], k0, kb);
            v16bf bl = load_frag(brl[t], k0, kb);
            acc[t] = __builtin_amdgcn_wmma_f32_16x16x32_bf16(false, ah, false, bh, (short)0, acc[t], false, false);
            acc[t] = __builtin_amdgcn_wmma_f32_16x16x32_bf16(false, ah, false, bl, (short)0, acc[t], false, false);
            acc[t] = __builtin_amdgcn_wmma_f32_16x16x32_bf16(false, al, false, bh, (short)0, acc[t], false, false);
        }
    }

    // C layout: lane L holds n = L%16, VGPR v holds m = v + 8*(L/16)
    const int mhalf = (lane >> 4) * 8;
#pragma unroll
    for (int t = 0; t < 4; ++t) {
        const int nG   = nT[t] * 16 + (lane & 15);
        const float nbv = nb[nG];
#pragma unroll
        for (int v = 0; v < 8; ++v) {
            int mG = mTile * 16 + mhalf + v;
            float d2 = fmaxf(na[mG] + nbv - 2.0f * acc[t][v], 0.0f);
            float val = d2;
            int   bi  = nG;
#pragma unroll
            for (int s = 1; s < 16; s <<= 1) {
                float ov = __shfl_xor(val, s, 32);
                int   oi = __shfl_xor(bi,  s, 32);
                if (ov < val || (ov == val && oi < bi)) { val = ov; bi = oi; }
            }
            if ((lane & 15) == 0) {
                unsigned long long packed =
                    ((unsigned long long)__float_as_uint(val) << 32) | (unsigned int)bi;
                atomicMin(rowmin + mG, packed);
            }
        }
    }
}

// ---------------------------------------------------------------------------
// Mismatch finalize: count = (ceil(d)-floor(d)) * (nn != idx), summed.
// ---------------------------------------------------------------------------
__global__ void __launch_bounds__(128) finalize_kernel(
    const unsigned long long* __restrict__ rowmin,
    const int* __restrict__ imap, int col,
    float* __restrict__ acc_slot, int K)
{
    int k = blockIdx.x * blockDim.x + threadIdx.x;
    if (k >= K) return;
    unsigned long long p = rowmin[k];
    float d2 = __uint_as_float((unsigned int)(p >> 32));
    int   nn = (int)(unsigned int)(p & 0xffffffffu);
    int  idx = imap[2 * k + col];
    float dist = sqrtf(d2);
    float c = (ceilf(dist) - floorf(dist)) * ((nn != idx) ? 1.0f : 0.0f);
    atomicAdd(acc_slot, c);
}

// ---------------------------------------------------------------------------
// Scalar combine.
// ---------------------------------------------------------------------------
__global__ void combine_kernel(const float* __restrict__ acc, float* __restrict__ out,
                               int N, int K)
{
    if (threadIdx.x == 0 && blockIdx.x == 0) {
        float invN = 1.0f / (float)N, invK = 1.0f / (float)K;
        out[0] = acc[0] * invN + acc[1] * invN +
                 acc[2] * invK + acc[4] * invK +
                 acc[3] * invK + acc[5] * invK;
    }
}

// ---------------------------------------------------------------------------
extern "C" void kernel_launch(void* const* d_in, const int* in_sizes, int n_in,
                              void* d_out, int out_size, void* d_ws, size_t ws_size,
                              hipStream_t stream) {
    const float* xw   = (const float*)d_in[0];
    const float* yw   = (const float*)d_in[1];
    const float* fw1  = (const float*)d_in[2];
    const float* fb1  = (const float*)d_in[3];
    const float* fw2  = (const float*)d_in[4];
    const float* fb2  = (const float*)d_in[5];
    const float* gw1  = (const float*)d_in[6];
    const float* gb1  = (const float*)d_in[7];
    const float* gw2  = (const float*)d_in[8];
    const float* gb2  = (const float*)d_in[9];
    const int*   imap = (const int*)d_in[10];

    const int XD = 100, YD = 120;
    const int N    = in_sizes[0] / XD;           // 25000
    const int K    = in_sizes[10] / 2;           // 8000
    const int NPAD = ((N + 15) / 16) * 16;       // 25008
    const int KPAD = ((K + 15) / 16) * 16;       // 8000
    const int NT   = NPAD / 16;                  // 1563
    const int MT   = KPAD / 16;                  // 500

    // carve workspace (~56 MB), 256B-aligned regions
    char* p = (char*)d_ws;
    auto carve = [&](size_t bytes) -> char* {
        char* q = p;
        p += (bytes + 255) & ~(size_t)255;
        return q;
    };
    float* acc        = (float*)carve(16 * sizeof(float));
    float* x_mapped   = (float*)carve((size_t)N * YD * 4);
    float* y_mapped   = (float*)carve((size_t)N * XD * 4);
    __bf16* Afx_hi    = (__bf16*)carve((size_t)KPAD * 128 * 2);
    __bf16* Afx_lo    = (__bf16*)carve((size_t)KPAD * 128 * 2);
    __bf16* Agy_hi    = (__bf16*)carve((size_t)KPAD * 128 * 2);
    __bf16* Agy_lo    = (__bf16*)carve((size_t)KPAD * 128 * 2);
    __bf16* Bfx_hi    = (__bf16*)carve((size_t)NPAD * 128 * 2);
    __bf16* Bfx_lo    = (__bf16*)carve((size_t)NPAD * 128 * 2);
    __bf16* Bgy_hi    = (__bf16*)carve((size_t)NPAD * 128 * 2);
    __bf16* Bgy_lo    = (__bf16*)carve((size_t)NPAD * 128 * 2);
    float* na_fx      = (float*)carve((size_t)KPAD * 4);
    float* na_gy      = (float*)carve((size_t)KPAD * 4);
    float* nb_fx      = (float*)carve((size_t)NPAD * 4);
    float* nb_gy      = (float*)carve((size_t)NPAD * 4);
    unsigned long long* rmin_fx = (unsigned long long*)carve((size_t)KPAD * 8);
    unsigned long long* rmin_gy = (unsigned long long*)carve((size_t)KPAD * 8);
    (void)ws_size;

    // 1) init accumulators + rowmins
    init_kernel<<<(KPAD + 127) / 128, 128, 0, stream>>>(acc, rmin_fx, rmin_gy, KPAD);

    // 2) fused MLP + round-trip + cycle norms
    mlp_cycle_kernel<<<(N + 127) / 128, 128, 0, stream>>>(
        xw, fw1, fb1, fw2, fb2, gw1, gb1, gw2, gb2,
        x_mapped, &acc[0], N, XD, YD);
    mlp_cycle_kernel<<<(N + 127) / 128, 128, 0, stream>>>(
        yw, gw1, gb1, gw2, gb2, fw1, fb1, fw2, fb2,
        y_mapped, &acc[1], N, YD, XD);

    // 3) split-bf16 conversion of targets (+ row norms)
    convert_B_kernel<<<(NPAD + 127) / 128, 128, 0, stream>>>(yw, N, YD, NPAD, Bfx_hi, Bfx_lo, nb_fx);
    convert_B_kernel<<<(NPAD + 127) / 128, 128, 0, stream>>>(xw, N, XD, NPAD, Bgy_hi, Bgy_lo, nb_gy);

    // 4) gather/convert A rows + supervised norms
    gather_A_kernel<<<(KPAD + 127) / 128, 128, 0, stream>>>(
        x_mapped, y_mapped, xw, yw, imap, K, KPAD,
        Afx_hi, Afx_lo, na_fx, Agy_hi, Agy_lo, na_gy, acc);

    // 5) WMMA 1-NN search (the ~88 GFLOP core): 8 waves/WG, 4 N-tiles/wave
    dim3 nnGrid(MT, (NT + 31) / 32);
    nn_min_kernel<<<nnGrid, 256, 0, stream>>>(Afx_hi, Afx_lo, Bfx_hi, Bfx_lo,
                                              na_fx, nb_fx, rmin_fx, NT);
    nn_min_kernel<<<nnGrid, 256, 0, stream>>>(Agy_hi, Agy_lo, Bgy_hi, Bgy_lo,
                                              na_gy, nb_gy, rmin_gy, NT);

    // 6) mismatch counts
    finalize_kernel<<<(K + 127) / 128, 128, 0, stream>>>(rmin_fx, imap, 0, &acc[4], K);
    finalize_kernel<<<(K + 127) / 128, 128, 0, stream>>>(rmin_gy, imap, 1, &acc[5], K);

    // 7) combine to scalar output
    combine_kernel<<<1, 32, 0, stream>>>(acc, (float*)d_out, N, K);
}
